// LSTM_80960133529703
// MI455X (gfx1250) — matrix operand order repaired and checked by reading, not compiled, not run on hardware
//
#include <hip/hip_runtime.h>
#include <hip/hip_bf16.h>

#define DEV __device__ __forceinline__
#define AS1 __attribute__((address_space(1)))
#define AS3 __attribute__((address_space(3)))

typedef _Float16 half16 __attribute__((ext_vector_type(16)));
typedef _Float16 half8  __attribute__((ext_vector_type(8)));
typedef float    f32x8  __attribute__((ext_vector_type(8)));
typedef int      v4i    __attribute__((ext_vector_type(4)));

constexpr int Bsz = 16384, T = 24, Fin = 7, H = 8;
constexpr int K1 = 192;          // T*H
constexpr int N1 = 1024, N2 = 2048, NC = 4;

// LDS B-tile: 128 rows x 32 halfs, padded row stride 40 halfs (80 B: still
// 16B-aligned, and rlo*20 dwords mod 64 = {0,4,...,60} -> conflict-free b128).
constexpr int LSTRIDE = 40;

#if __has_builtin(__builtin_amdgcn_global_load_async_to_lds_b128)
#define HAVE_ASYNC 1
#else
#define HAVE_ASYNC 0
#endif

#if __has_builtin(__builtin_amdgcn_s_wait_asynccnt)
#define WAIT_ASYNC() __builtin_amdgcn_s_wait_asynccnt(0)
#elif HAVE_ASYNC
#define WAIT_ASYNC() asm volatile("s_wait_asynccnt 0x0" ::: "memory")
#else
#define WAIT_ASYNC() ((void)0)
#endif

#if __has_builtin(__builtin_amdgcn_sched_group_barrier)
// Keep the inner block shaped as: 16 ds_load_b128 first, then 8 WMMAs, so
// s_wait_dscnt become partial (DS returns in order) instead of 0 per WMMA.
#define SCHED_DS_THEN_WMMA()                          \
  do {                                                \
    __builtin_amdgcn_sched_group_barrier(0x100, 16, 0); /* DS reads */ \
    __builtin_amdgcn_sched_group_barrier(0x008, 8, 0);  /* MFMA/WMMA */ \
  } while (0)
#else
#define SCHED_DS_THEN_WMMA() ((void)0)
#endif

// ---------------------------------------------------------------- LSTM ----
DEV float sigm(float v) { return 1.0f / (1.0f + __expf(-v)); }

template <int IN>
DEV void lstm_cell(const float* __restrict__ wih, const float* __restrict__ whh,
                   const float* __restrict__ bias, const float* in,
                   float* h, float* c) {
  float z[4 * H];
#pragma unroll
  for (int g = 0; g < 4 * H; ++g) {
    float acc = bias[g];
#pragma unroll
    for (int f = 0; f < IN; ++f) acc += wih[g * IN + f] * in[f];
#pragma unroll
    for (int j = 0; j < H; ++j) acc += whh[g * H + j] * h[j];
    z[g] = acc;
  }
#pragma unroll
  for (int j = 0; j < H; ++j) {
    float ig = sigm(z[j]);
    float fg = sigm(z[H + j]);
    float gg = tanhf(z[2 * H + j]);
    float og = sigm(z[3 * H + j]);
    c[j] = fg * c[j] + ig * gg;
    h[j] = og * tanhf(c[j]);
  }
}

// LDS layout (floats): [0,224) Wih0 | [224,480) Whh0 | [480,512) b0
// then 3 blocks of 544: [0,256) Wih | [256,512) Whh | [512,544) bias
__global__ __launch_bounds__(256) void lstm_kernel(
    const float* __restrict__ x,
    const float* __restrict__ Wih0, const float* __restrict__ Whh0,
    const float* __restrict__ bih0, const float* __restrict__ bhh0,
    const float* __restrict__ Wih1, const float* __restrict__ Whh1,
    const float* __restrict__ bih1, const float* __restrict__ bhh1,
    const float* __restrict__ Wih2, const float* __restrict__ Whh2,
    const float* __restrict__ bih2, const float* __restrict__ bhh2,
    const float* __restrict__ Wih3, const float* __restrict__ Whh3,
    const float* __restrict__ bih3, const float* __restrict__ bhh3,
    _Float16* __restrict__ act0) {
  __shared__ float s[2144];
  const int tid = threadIdx.x;

  for (int i = tid; i < 224; i += 256) s[i] = Wih0[i];
  for (int i = tid; i < 256; i += 256) s[224 + i] = Whh0[i];
  if (tid < 32) s[480 + tid] = bih0[tid] + bhh0[tid];
  {
    const float* wi[3] = {Wih1, Wih2, Wih3};
    const float* wh[3] = {Whh1, Whh2, Whh3};
    const float* bi[3] = {bih1, bih2, bih3};
    const float* bh[3] = {bhh1, bhh2, bhh3};
#pragma unroll
    for (int l = 0; l < 3; ++l) {
      const int base = 512 + l * 544;
      for (int i = tid; i < 256; i += 256) {
        s[base + i] = wi[l][i];
        s[base + 256 + i] = wh[l][i];
      }
      if (tid < 32) s[base + 512 + tid] = bi[l][tid] + bh[l][tid];
    }
  }
  __syncthreads();

  const int b = blockIdx.x * 256 + tid;

  float h0[H] = {}, c0[H] = {}, h1[H] = {}, c1[H] = {};
  float h2[H] = {}, c2[H] = {}, h3[H] = {}, c3[H] = {};
  const float* xb = x + (size_t)b * T * Fin;

  for (int t = 0; t < T; ++t) {
    float in[Fin];
#pragma unroll
    for (int f = 0; f < Fin; ++f) in[f] = xb[t * Fin + f];

    lstm_cell<Fin>(&s[0], &s[224], &s[480], in, h0, c0);
    lstm_cell<H>(&s[512], &s[512 + 256], &s[512 + 512], h0, h1, c1);
    lstm_cell<H>(&s[1056], &s[1056 + 256], &s[1056 + 512], h1, h2, c2);
    lstm_cell<H>(&s[1600], &s[1600 + 256], &s[1600 + 512], h2, h3, c3);

    half8 hv;
#pragma unroll
    for (int j = 0; j < H; ++j) hv[j] = (_Float16)h3[j];
    *(half8*)(act0 + (size_t)b * K1 + t * H) = hv;  // 16B store
  }
}

// ---------------------------------------------------------------- GEMM ----
// CDNA5 wave32 WMMA fragment builders (ISA 7.12.2 layouts).
// A (16x32 f16): lane rlo=l&15 -> row; hi=l>>4 selects K sub-block.
DEV half16 load_a_frag(const _Float16* __restrict__ A, int row, int K, int k0,
                       int hi) {
  const _Float16* p = A + (size_t)row * K + k0 + hi * 8;
  half8 lo = *(const half8*)p;
  half8 hh = *(const half8*)(p + 16);
  return __builtin_shufflevector(lo, hh, 0, 1, 2, 3, 4, 5, 6, 7, 8, 9, 10, 11,
                                 12, 13, 14, 15);
}
// B (32x16 f16) from LDS tile [128][LSTRIDE]: lane rlo -> col; K = hi*8+0..7, 16+hi*8+0..7
DEV half16 lds_b_frag(const _Float16* sbuf, int n_local, int hi) {
  const _Float16* p = sbuf + n_local * LSTRIDE + hi * 8;
  half8 lo = *(const half8*)p;
  half8 hh = *(const half8*)(p + 16);
  return __builtin_shufflevector(lo, hh, 0, 1, 2, 3, 4, 5, 6, 7, 8, 9, 10, 11,
                                 12, 13, 14, 15);
}

// Stage one 128x32 f16 B tile into LDS. 512 16B chunks, 2 per thread.
DEV void stage_b(const _Float16* __restrict__ W, _Float16* sbuf, int nj,
                 int k0, int K, int tid) {
#pragma unroll
  for (int c = tid; c < 512; c += 256) {
    const int n = c >> 2;            // 0..127
    const int kk = (c & 3) * 8;      // 0,8,16,24
    const _Float16* g = W + (size_t)(nj * 128 + n) * K + k0 + kk;
    _Float16* l = sbuf + n * LSTRIDE + kk;
#if HAVE_ASYNC
    __builtin_amdgcn_global_load_async_to_lds_b128((AS1 v4i*)g, (AS3 v4i*)l,
                                                   0, 0);
#else
    *(half8*)l = *(const half8*)g;   // load_b128 + ds_store_b128
#endif
  }
}

// Block (8 waves) computes 128 rows x 128 cols; B tile staged in LDS once per
// k-step (async, double-buffered). A fragment register double-buffered; all
// 16 ds_load_b128 issued before the 8 WMMAs so waits are partial.
// Grid: x = N/128, y = M/128 ; block = 256.
__global__ __launch_bounds__(256) void gemm_bias_relu_f16(
    const _Float16* __restrict__ A,   // [M,K] f16
    const _Float16* __restrict__ W,   // [N,K] f16 (i.e. W.T of the [K,N] op)
    const float* __restrict__ bias,   // [N]
    _Float16* __restrict__ C,         // [M,N] f16 (relu'd)
    int M, int N, int K) {
  __shared__ _Float16 sb[2][128 * LSTRIDE];

  const int tid = threadIdx.x;
  const int lane = tid & 31;
  const int wib = tid >> 5;
  const int mi = blockIdx.y * 8 + wib;  // 16-row strip index
  const int nj = blockIdx.x;            // 128-col group

  const int rlo = lane & 15;
  const int hi = lane >> 4;
  const int row = mi * 16 + rlo;

  f32x8 acc[8] = {};
  const int KS = K >> 5;

  stage_b(W, &sb[0][0], nj, 0, K, tid);
  half16 a = load_a_frag(A, row, K, 0, hi);

  for (int ks = 0; ks < KS; ++ks) {
    WAIT_ASYNC();          // my async writes to sb[ks&1] complete
    __syncthreads();       // everyone's writes visible; prev compute done

    // Stage B for k-step ks+1 (async -> LDS) and prefetch A for ks+1 into
    // registers; both land while the 8 WMMAs below execute.
    half16 a_next = a;
    if (ks + 1 < KS) {
      stage_b(W, &sb[(ks + 1) & 1][0], nj, (ks + 1) * 32, K, tid);
      a_next = load_a_frag(A, row, K, (ks + 1) * 32, hi);
    }

    const _Float16* sbuf = &sb[ks & 1][0];
    half16 bfr[8];
#pragma unroll
    for (int t = 0; t < 8; ++t)  // 16 ds_load_b128 issued back-to-back
      bfr[t] = lds_b_frag(sbuf, t * 16 + rlo, hi);
#pragma unroll
    for (int t = 0; t < 8; ++t)  // waits on DS loads amortize across WMMAs
      acc[t] = __builtin_amdgcn_wmma_f32_16x16x32_f16(
          false, a, false, bfr[t], (short)0, acc[t], false, false);
    SCHED_DS_THEN_WMMA();

    a = a_next;
  }

#pragma unroll
  for (int t = 0; t < 8; ++t) {
    const int n = nj * 128 + t * 16 + rlo;
    const float bv = bias[n];
#pragma unroll
    for (int r = 0; r < 8; ++r) {
      const int m = mi * 16 + r + hi * 8;
      float v = acc[t][r] + bv;
      v = v > 0.0f ? v : 0.0f;  // relu
      C[(size_t)m * N + n] = (_Float16)v;
    }
  }
}

// -------------------------------------------------------------- dense3 ----
__global__ __launch_bounds__(256) void dense3_kernel(
    const _Float16* __restrict__ act2, const float* __restrict__ W,
    const float* __restrict__ bias, float* __restrict__ out) {
  const int b = blockIdx.x * 256 + threadIdx.x;
  const _Float16* a = act2 + (size_t)b * N2;
  float acc[NC] = {};
  for (int k = 0; k < N2; k += 8) {
    half8 av8 = *(const half8*)(a + k);
#pragma unroll
    for (int u = 0; u < 8; ++u) {
      const float av = (float)av8[u];
#pragma unroll
      for (int j = 0; j < NC; ++j) acc[j] += av * W[j * N2 + k + u];
    }
  }
#pragma unroll
  for (int j = 0; j < NC; ++j) out[b * NC + j] = acc[j] + bias[j];
}

// ------------------------------------------------------------- convert ----
__global__ __launch_bounds__(256) void f32_to_f16(const float* __restrict__ s,
                                                  _Float16* __restrict__ d,
                                                  int n) {
  const int i = blockIdx.x * 256 + threadIdx.x;
  if (i < n) d[i] = (_Float16)s[i];
}

// -------------------------------------------------------------- launch ----
extern "C" void kernel_launch(void* const* d_in, const int* in_sizes, int n_in,
                              void* d_out, int out_size, void* d_ws,
                              size_t ws_size, hipStream_t stream) {
  const float* x = (const float*)d_in[0];
  const float* Wih0 = (const float*)d_in[1];
  const float* Whh0 = (const float*)d_in[2];
  const float* bih0 = (const float*)d_in[3];
  const float* bhh0 = (const float*)d_in[4];
  const float* Wih1 = (const float*)d_in[5];
  const float* Whh1 = (const float*)d_in[6];
  const float* bih1 = (const float*)d_in[7];
  const float* bhh1 = (const float*)d_in[8];
  const float* Wih2 = (const float*)d_in[9];
  const float* Whh2 = (const float*)d_in[10];
  const float* bih2 = (const float*)d_in[11];
  const float* bhh2 = (const float*)d_in[12];
  const float* Wih3 = (const float*)d_in[13];
  const float* Whh3 = (const float*)d_in[14];
  const float* bih3 = (const float*)d_in[15];
  const float* bhh3 = (const float*)d_in[16];
  const float* Wd1 = (const float*)d_in[17];
  const float* bd1 = (const float*)d_in[18];
  const float* Wd2 = (const float*)d_in[19];
  const float* bd2 = (const float*)d_in[20];
  const float* Wd3 = (const float*)d_in[21];
  const float* bd3 = (const float*)d_in[22];

  char* ws = (char*)d_ws;
  const size_t off_act0 = 0;                                // B*192 f16
  const size_t off_act1 = off_act0 + (size_t)Bsz * K1 * 2;  // B*1024 f16
  const size_t off_act2 = off_act1 + (size_t)Bsz * N1 * 2;  // B*2048 f16
  const size_t off_w1h = off_act2 + (size_t)Bsz * N2 * 2;   // 1024*192 f16
  const size_t off_w2h = off_w1h + (size_t)N1 * K1 * 2;     // 2048*1024 f16
  _Float16* act0 = (_Float16*)(ws + off_act0);
  _Float16* act1 = (_Float16*)(ws + off_act1);
  _Float16* act2 = (_Float16*)(ws + off_act2);
  _Float16* W1h = (_Float16*)(ws + off_w1h);
  _Float16* W2h = (_Float16*)(ws + off_w2h);

  // Convert dense weights to f16 (stored [N,K] row-major, same as reference).
  f32_to_f16<<<(N1 * K1 + 255) / 256, 256, 0, stream>>>(Wd1, W1h, N1 * K1);
  f32_to_f16<<<(N2 * N1 + 255) / 256, 256, 0, stream>>>(Wd2, W2h, N2 * N1);

  // Fused 4-layer LSTM, one thread per batch element.
  lstm_kernel<<<Bsz / 256, 256, 0, stream>>>(
      x, Wih0, Whh0, bih0, bhh0, Wih1, Whh1, bih1, bhh1, Wih2, Whh2, bih2,
      bhh2, Wih3, Whh3, bih3, bhh3, act0);

  // Dense1: [B,192] x [192,1024] + b1, relu
  {
    dim3 grid(N1 / 128, Bsz / 128);
    gemm_bias_relu_f16<<<grid, 256, 0, stream>>>(act0, W1h, bd1, act1, Bsz, N1,
                                                 K1);
  }
  // Dense2: [B,1024] x [1024,2048] + b2, relu (dominant: ~69 GFLOP)
  {
    dim3 grid(N2 / 128, Bsz / 128);
    gemm_bias_relu_f16<<<grid, 256, 0, stream>>>(act1, W2h, bd2, act2, Bsz, N2,
                                                 N1);
  }
  // Dense3: [B,2048] x [2048,4] + b3 (f32, lane-uniform weights)
  dense3_kernel<<<Bsz / 256, 256, 0, stream>>>(act2, Wd3, bd3, (float*)d_out);
}
